// MultiScaleSSMBank_42279658062148
// MI455X (gfx1250) — matrix operand order repaired and checked by reading, not compiled, not run on hardware
//
#include <hip/hip_runtime.h>
#include <hip/hip_fp16.h>

// MultiScaleSSMBank for MI455X (gfx1250, wave32).
// Phases:
//  1. pack:   xm = x*mask (f32 for skip path) + f16 A-fragments of xm; pack Bw/Cw B-fragments.
//  2. gemm_xb: XB[s,b,t,:] = xm[b,t,:] @ Bw[s].T via v_wmma_f32_16x16x32_f16.
//  3. scan:   one wave per (s,b); per step: dt-MLP (LDS h-broadcast + shfl reduce),
//             AR update of h; h_t stored directly as packed f16 A-fragment.
//  4. gemm_y: Y = H@Cw.T + Dp*xm, fused layernorm, write all_outputs (WMMA).
//  5. gates:  silu(concat@gW1.T)@gW2.T -> sigmoid -> mixed.

#define NB_B 32
#define NB_T 1024
#define NB_D 128
#define NB_H 32
#define NB_S 8
#define LN_EPSF 1e-5f
#define ROWS (NB_B * NB_T)      // 32768 rows (b*T + t)
#define RT_TILES (ROWS / 16)    // 2048 row tiles

typedef __attribute__((ext_vector_type(16))) _Float16 v16h;
typedef __attribute__((ext_vector_type(8)))  float    v8f;

// A-matrix (16x32 f16) per-lane K index for element e (ISA 7.12.2 table):
// bits: [2:0] = pair-structure, [3] = lane-half*8, [4] = e-high*16
__device__ __forceinline__ int ka_idx(int lane, int e) {
    return (((e >> 3) & 1) << 4) | ((lane >> 4) << 3) |
           ((((e >> 1) & 3) << 1) | (e & 1));
}
// B-matrix (32x16 f16): same K order as A (table order)
__device__ __forceinline__ int kb_idx(int lane, int e) {
    return (((e >> 3) & 1) << 4) | ((lane >> 4) << 3) | (e & 7);
}

// ---------------------------------------------------------------- phase 1a
__global__ void pack_x_kernel(const float* __restrict__ x,
                              const float* __restrict__ mask,
                              float* __restrict__ xm,
                              _Float16* __restrict__ pa1) {
    int gtid = blockIdx.x * blockDim.x + threadIdx.x;   // RT_TILES*4 frags * 32 lanes
    int lane = gtid & 31;
    int f    = gtid >> 5;             // fragment id = rt*4 + kc
    int rt   = f >> 2, kc = f & 3;
    int row  = rt * 16 + (lane & 15); // row = b*T + t
    const float* xr = x    + (size_t)row * NB_D;
    const float* mr = mask + (size_t)row * NB_D;
    float*      xmr = xm   + (size_t)row * NB_D;
    _Float16*   dst = pa1  + (((size_t)f * 32) + lane) * 16;
#pragma unroll
    for (int e = 0; e < 16; ++e) {
        int d = kc * 32 + ka_idx(lane, e);
        float v = xr[d] * mr[d];
        xmr[d]  = v;                 // each (row,d) covered exactly once
        dst[e]  = (_Float16)v;
    }
}

// ---------------------------------------------------------------- phase 1b
__global__ void pack_w_kernel(const float* __restrict__ Bw,
                              const float* __restrict__ Cw,
                              _Float16* __restrict__ pbB,
                              _Float16* __restrict__ pcC) {
    int gtid = blockIdx.x * blockDim.x + threadIdx.x;   // 128 frags * 32
    int lane = gtid & 31;
    int f    = gtid >> 5;
    if (f < 64) {   // Bw fragment: f = (s*4 + kc)*2 + nt ; B[k,n] = Bw[s,n,k]
        int s  = f >> 3, kc = (f >> 1) & 3, nt = f & 1;
        int n  = nt * 16 + (lane & 15);
        _Float16* dst = pbB + (((size_t)f * 32) + lane) * 16;
#pragma unroll
        for (int e = 0; e < 16; ++e) {
            int k = kc * 32 + kb_idx(lane, e);
            dst[e] = (_Float16)Bw[((size_t)s * NB_H + n) * NB_D + k];
        }
    } else {        // Cw fragment: f2 = s*8 + nt ; B[k,n] = Cw[s,n(d),k(h)]
        int f2 = f - 64;
        int s  = f2 >> 3, nt = f2 & 7;
        int n  = nt * 16 + (lane & 15);
        _Float16* dst = pcC + (((size_t)f2 * 32) + lane) * 16;
#pragma unroll
        for (int e = 0; e < 16; ++e) {
            int k = kb_idx(lane, e);
            dst[e] = (_Float16)Cw[((size_t)s * NB_D + n) * NB_H + k];
        }
    }
}

// ---------------------------------------------------------------- phase 2
__global__ void gemm_xb_kernel(const _Float16* __restrict__ pa1,
                               const _Float16* __restrict__ pbB,
                               float* __restrict__ XB) {
    int wid  = (blockIdx.x * blockDim.x + threadIdx.x) >> 5;
    int lane = threadIdx.x & 31;
    int s    = wid / RT_TILES;
    int rt   = wid % RT_TILES;
    v8f c0 = {}; v8f c1 = {};
#pragma unroll
    for (int kc = 0; kc < 4; ++kc) {
        const _Float16* ap = pa1 + ((((size_t)rt * 4 + kc) * 32) + lane) * 16;
        if (kc < 3) __builtin_prefetch(ap + 32 * 16, 0, 0);   // global_prefetch_b8
        v16h a  = *(const v16h*)ap;
        v16h b0 = *(const v16h*)(pbB + ((((size_t)(s * 4 + kc) * 2 + 0) * 32) + lane) * 16);
        v16h b1 = *(const v16h*)(pbB + ((((size_t)(s * 4 + kc) * 2 + 1) * 32) + lane) * 16);
        c0 = __builtin_amdgcn_wmma_f32_16x16x32_f16(false, a, false, b0, (short)0, c0, false, false);
        c1 = __builtin_amdgcn_wmma_f32_16x16x32_f16(false, a, false, b1, (short)0, c1, false, false);
    }
    int mbase = (lane >> 4) << 3;
    int n     = lane & 15;
#pragma unroll
    for (int g = 0; g < 8; ++g) {
        int row = rt * 16 + mbase + g;
        int b   = row >> 10, t = row & 1023;
        size_t base = (((size_t)(s * NB_B + b)) * NB_T + t) * NB_H;
        XB[base + n]      = c0[g];
        XB[base + 16 + n] = c1[g];
    }
}

// ---------------------------------------------------------------- phase 3
__global__ __launch_bounds__(32) void scan_kernel(
        const float* __restrict__ delta_ts, const float* __restrict__ XB,
        const float* __restrict__ log_neg_A, const float* __restrict__ W1,
        const float* __restrict__ b1, const float* __restrict__ W2,
        const float* __restrict__ b2, _Float16* __restrict__ pa3,
        float* __restrict__ hfin) {
    int s = blockIdx.x >> 5;
    int b = blockIdx.x & 31;
    int j = threadIdx.x;                 // lane owns h_j
    float w1h[NB_H];
    const float* w1row = W1 + ((size_t)s * NB_H + j) * (NB_H + 1);
    float w1g = w1row[0];
#pragma unroll
    for (int k = 0; k < NB_H; ++k) w1h[k] = w1row[1 + k];
    float b1j = b1[s * NB_H + j];
    float w2j = W2[s * NB_H + j];
    float b2s = b2[s];
    float Ad  = -__expf(log_neg_A[s * NB_H + j]);
    __shared__ float sh[NB_H];
    float h = 0.f;
    const float* xbp = XB + (((size_t)(s * NB_B + b)) * NB_T) * NB_H + j;
    const float* gp  = delta_ts + (size_t)b * NB_T;
    size_t pa3_s  = (size_t)s * RT_TILES * 512;          // halfs per scale
    int    e_out  = (j & 7) | ((j >> 4) << 3);           // packed-A element for K=j
    int    lane_h = ((j >> 3) & 1) << 4;                 // packed-A lane-half for K=j
    for (int t = 0; t < NB_T; ++t) {
        float gap = gp[t];
        float xb  = xbp[(size_t)t * NB_H];
        sh[j] = h;
        __syncthreads();
        float u0 = b1j + w1g * gap, u1 = 0.f, u2 = 0.f, u3 = 0.f;
#pragma unroll
        for (int k = 0; k < NB_H; k += 4) {
            u0 += w1h[k]     * sh[k];
            u1 += w1h[k + 1] * sh[k + 1];
            u2 += w1h[k + 2] * sh[k + 2];
            u3 += w1h[k + 3] * sh[k + 3];
        }
        __syncthreads();
        float u  = (u0 + u1) + (u2 + u3);
        float su = u / (1.f + __expf(-u));               // silu
        float v  = su * w2j;
#pragma unroll
        for (int off = 16; off > 0; off >>= 1) v += __shfl_xor(v, off, 32);
        float a0  = v + b2s;
        float dtv = (a0 > 20.f) ? a0 : log1pf(__expf(a0));   // softplus
        float ex  = fminf(fmaxf(Ad * dtv, -20.f), 0.f);
        float ab  = __expf(ex);
        h = fminf(fmaxf(ab * h + xb * dtv, -50.f), 50.f);
        int tile = (b * NB_T + t) >> 4;
        pa3[pa3_s + (((size_t)tile * 32) + ((t & 15) | lane_h)) * 16 + e_out] = (_Float16)h;
    }
    hfin[((size_t)s * NB_B + b) * NB_H + j] = h;
}

// ---------------------------------------------------------------- phase 4
__global__ void gemm_y_kernel(const _Float16* __restrict__ pa3,
                              const _Float16* __restrict__ pcC,
                              const float* __restrict__ xm,
                              const float* __restrict__ Dp,
                              const float* __restrict__ ln_g,
                              const float* __restrict__ ln_b,
                              float* __restrict__ out_all) {
    int wid  = (blockIdx.x * blockDim.x + threadIdx.x) >> 5;
    int lane = threadIdx.x & 31;
    int s    = wid / RT_TILES;
    int rt   = wid % RT_TILES;
    v16h a = *(const v16h*)(pa3 + ((((size_t)s * RT_TILES + rt) * 32) + lane) * 16);
    v8f c[8];
#pragma unroll
    for (int nt = 0; nt < 8; ++nt) {
        v16h bb = *(const v16h*)(pcC + ((((size_t)s * 8 + nt) * 32) + lane) * 16);
        v8f z = {};
        c[nt] = __builtin_amdgcn_wmma_f32_16x16x32_f16(false, a, false, bb, (short)0, z, false, false);
    }
    int mbase = (lane >> 4) << 3;
    int n     = lane & 15;
#pragma unroll
    for (int nt = 0; nt < 8; ++nt) {        // skip connection Dp*x
        int d = nt * 16 + n;
        float dp = Dp[s * NB_D + d];
#pragma unroll
        for (int g = 0; g < 8; ++g) {
            int row = rt * 16 + mbase + g;
            c[nt][g] += dp * xm[(size_t)row * NB_D + d];
        }
    }
    float mu[8], rs[8];
#pragma unroll
    for (int g = 0; g < 8; ++g) {           // mean over D=128 (8 vals/lane x 16 lanes)
        float sum = 0.f;
#pragma unroll
        for (int nt = 0; nt < 8; ++nt) sum += c[nt][g];
#pragma unroll
        for (int off = 8; off > 0; off >>= 1) sum += __shfl_xor(sum, off, 32);
        mu[g] = sum * (1.f / NB_D);
    }
#pragma unroll
    for (int g = 0; g < 8; ++g) {           // variance
        float sq = 0.f;
#pragma unroll
        for (int nt = 0; nt < 8; ++nt) { float dv = c[nt][g] - mu[g]; sq += dv * dv; }
#pragma unroll
        for (int off = 8; off > 0; off >>= 1) sq += __shfl_xor(sq, off, 32);
        rs[g] = rsqrtf(sq * (1.f / NB_D) + LN_EPSF);
    }
#pragma unroll
    for (int nt = 0; nt < 8; ++nt) {
        int d = nt * 16 + n;
        float gg = ln_g[s * NB_D + d], bb = ln_b[s * NB_D + d];
#pragma unroll
        for (int g = 0; g < 8; ++g) {
            int row = rt * 16 + mbase + g;
            int b2  = row >> 10, tt = row & 1023;
            out_all[(((size_t)(s * NB_B + b2)) * NB_T + tt) * NB_D + d] =
                (c[nt][g] - mu[g]) * rs[g] * gg + bb;
        }
    }
}

// ---------------------------------------------------------------- phase 5
__global__ void gates1_kernel(const float* __restrict__ out_all,
                              const float* __restrict__ gW1,
                              const float* __restrict__ gb1,
                              float* __restrict__ hid) {
    int gtid = blockIdx.x * blockDim.x + threadIdx.x;   // 32768 = B * (S*D)
    int b  = gtid >> 10;
    int jj = gtid & 1023;
    const float* w = gW1 + (size_t)jj * (NB_S * NB_D);
    float acc = gb1[jj];
    for (int s = 0; s < NB_S; ++s) {
        const float* emb = out_all + (((size_t)(s * NB_B + b)) * NB_T + (NB_T - 1)) * NB_D;
#pragma unroll 4
        for (int d = 0; d < NB_D; ++d) acc += emb[d] * w[s * NB_D + d];
    }
    hid[(size_t)b * 1024 + jj] = acc / (1.f + __expf(-acc));   // silu
}

__global__ void gates2_kernel(const float* __restrict__ hid,
                              const float* __restrict__ gW2,
                              const float* __restrict__ gb2,
                              const float* __restrict__ out_all,
                              float* __restrict__ mixed) {
    __shared__ float gts[NB_B][NB_S];
    int tid = threadIdx.x;
    int b = tid >> 3, s = tid & 7;
    const float* hb = hid + (size_t)b * 1024;
    const float* w  = gW2 + (size_t)s * 1024;
    float acc = gb2[s];
    for (int j = 0; j < 1024; ++j) acc += hb[j] * w[j];
    gts[b][s] = 1.f / (1.f + __expf(-acc));
    __syncthreads();
#pragma unroll
    for (int m = 0; m < 16; ++m) {
        int o  = tid * 16 + m;          // 0..4095 over [B,D]
        int b2 = o >> 7, d = o & 127;
        float acc2 = 0.f;
#pragma unroll
        for (int s2 = 0; s2 < NB_S; ++s2)
            acc2 += out_all[(((size_t)(s2 * NB_B + b2)) * NB_T + (NB_T - 1)) * NB_D + d] * gts[b2][s2];
        mixed[o] = acc2;
    }
}

// ---------------------------------------------------------------- launch
extern "C" void kernel_launch(void* const* d_in, const int* in_sizes, int n_in,
                              void* d_out, int out_size, void* d_ws, size_t ws_size,
                              hipStream_t stream) {
    (void)in_sizes; (void)n_in; (void)out_size; (void)ws_size;
    const float* x         = (const float*)d_in[0];
    const float* delta_ts  = (const float*)d_in[1];
    const float* mask      = (const float*)d_in[2];
    const float* log_neg_A = (const float*)d_in[3];
    const float* Bw        = (const float*)d_in[4];
    const float* Cw        = (const float*)d_in[5];
    const float* Dp        = (const float*)d_in[6];
    const float* W1        = (const float*)d_in[7];
    const float* b1        = (const float*)d_in[8];
    const float* W2        = (const float*)d_in[9];
    const float* b2        = (const float*)d_in[10];
    const float* ln_g      = (const float*)d_in[11];
    const float* ln_b      = (const float*)d_in[12];
    const float* gW1       = (const float*)d_in[13];
    const float* gb1       = (const float*)d_in[14];
    const float* gW2       = (const float*)d_in[15];
    const float* gb2       = (const float*)d_in[16];

    float* out     = (float*)d_out;
    float* mixed   = out;                                         // [B,D]
    float* out_all = out + (size_t)NB_B * NB_D;                   // [S,B,T,D]
    float* hfin    = out_all + (size_t)NB_S * NB_B * NB_T * NB_D; // [S,B,H]

    // workspace layout (~72.5 MB total)
    char* ws = (char*)d_ws;
    float*    xm  = (float*)ws;    ws += (size_t)ROWS * NB_D * sizeof(float);        // 16 MB
    float*    XB  = (float*)ws;    ws += (size_t)NB_S * ROWS * NB_H * sizeof(float); // 32 MB
    float*    hid = (float*)ws;    ws += (size_t)NB_B * 1024 * sizeof(float);        // 128 KB
    _Float16* pa1 = (_Float16*)ws; ws += (size_t)ROWS * NB_D * 2;                    // 8 MB
    _Float16* pa3 = (_Float16*)ws; ws += (size_t)NB_S * ROWS * NB_H * 2;             // 16 MB
    _Float16* pbB = (_Float16*)ws; ws += (size_t)64 * 32 * 16 * 2;                   // 64 KB
    _Float16* pcC = (_Float16*)ws; ws += (size_t)64 * 32 * 16 * 2;                   // 64 KB

    pack_x_kernel <<< (RT_TILES * 4 * 32) / 256, 256, 0, stream >>>(x, mask, xm, pa1);
    pack_w_kernel <<< 16, 256, 0, stream >>>(Bw, Cw, pbB, pcC);
    gemm_xb_kernel<<< (NB_S * RT_TILES) / 8, 256, 0, stream >>>(pa1, pbB, XB);
    scan_kernel   <<< NB_S * NB_B, 32, 0, stream >>>(delta_ts, XB, log_neg_A,
                                                     W1, b1, W2, b2, pa3, hfin);
    gemm_y_kernel <<< (NB_S * RT_TILES) / 8, 256, 0, stream >>>(pa3, pcC, xm, Dp,
                                                                ln_g, ln_b, out_all);
    gates1_kernel <<< (NB_B * 1024) / 256, 256, 0, stream >>>(out_all, gW1, gb1, hid);
    gates2_kernel <<< 1, 256, 0, stream >>>(hid, gW2, gb2, out_all, mixed);
}
